// MoELayer_52355651338921
// MI455X (gfx1250) — matrix operand order repaired and checked by reading, not compiled, not run on hardware
//
#include <hip/hip_runtime.h>
#include <hip/hip_bf16.h>

#define N_TOK 8192
#define DIM   1024
#define NEXP  8
#define HID   4096

#define TM 128   // rows per workgroup tile
#define TN 128   // cols per workgroup tile
#define TK 32    // k-step (one bf16 WMMA depth)
#define LSTR 40  // LDS row stride in halves (80B: 16B-aligned, bank-conflict-friendly)
#define PIPE 3   // async pipeline depth
#define TILE_BYTES (TM * LSTR * 2)

typedef __attribute__((ext_vector_type(16))) __bf16 v16bf;
typedef __attribute__((ext_vector_type(8)))  float  v8f;

static __device__ __forceinline__ unsigned short f2bf(float f) {
  union { float f; unsigned u; } v; v.f = f;
  unsigned r = v.u + 0x7fffu + ((v.u >> 16) & 1u);   // round-to-nearest-even
  return (unsigned short)(r >> 16);
}
static __device__ __forceinline__ unsigned pk2(float a, float b) {
  return (unsigned)f2bf(a) | ((unsigned)f2bf(b) << 16);
}

// CDNA5 async global->LDS copy, 16B per lane, tracked by ASYNCcnt.
#define ASYNC_CP_B128(ldsAddr, gAddr)                                        \
  asm volatile("global_load_async_to_lds_b128 %0, %1, off"                   \
               :: "v"(ldsAddr), "v"(gAddr) : "memory")
#define WAIT_ASYNC0() asm volatile("s_wait_asynccnt 0x0" ::: "memory")
#define WAIT_ASYNC4() asm volatile("s_wait_asynccnt 0x4" ::: "memory")

// ---------------------------------------------------------------- gate + argmax
__global__ __launch_bounds__(256) void gate_kernel(
    const float* __restrict__ x, const float* __restrict__ gw,
    const float* __restrict__ gb, int* __restrict__ top1)
{
  const int n    = blockIdx.x * 8 + (threadIdx.x >> 5);  // one wave per token
  const int lane = threadIdx.x & 31;
  float acc[NEXP];
#pragma unroll
  for (int e = 0; e < NEXP; ++e) acc[e] = 0.0f;
  const float* xr = x + (size_t)n * DIM;
  for (int d = lane; d < DIM; d += 32) {
    const float xv = xr[d];
    const float4 a = *(const float4*)(gw + d * NEXP);
    const float4 b = *(const float4*)(gw + d * NEXP + 4);
    acc[0] += xv * a.x; acc[1] += xv * a.y; acc[2] += xv * a.z; acc[3] += xv * a.w;
    acc[4] += xv * b.x; acc[5] += xv * b.y; acc[6] += xv * b.z; acc[7] += xv * b.w;
  }
#pragma unroll
  for (int off = 16; off > 0; off >>= 1)
#pragma unroll
    for (int e = 0; e < NEXP; ++e) acc[e] += __shfl_down(acc[e], off, 32);
  if (lane == 0) {
    float best = acc[0] + gb[0]; int bi = 0;
#pragma unroll
    for (int e = 1; e < NEXP; ++e) {
      const float v = acc[e] + gb[e];
      if (v > best) { best = v; bi = e; }   // strict '>' keeps first max (jnp.argmax)
    }
    top1[n] = bi;
  }
}

// ---------------------------------------------------------------- bucketing
__global__ void hist_kernel(const int* __restrict__ top1, int* __restrict__ counts) {
  const int n = blockIdx.x * blockDim.x + threadIdx.x;
  if (n < N_TOK) atomicAdd(&counts[top1[n]], 1);
}
__global__ void scan_kernel(const int* __restrict__ counts, int* __restrict__ offs,
                            int* __restrict__ cursor) {
  if (threadIdx.x == 0) {
    int s = 0;
    for (int e = 0; e < NEXP; ++e) { offs[e] = s; cursor[e] = s; s += counts[e]; }
    offs[NEXP] = s;
  }
}
__global__ void scatter_kernel(const int* __restrict__ top1, int* __restrict__ cursor,
                               int* __restrict__ perm) {
  const int n = blockIdx.x * blockDim.x + threadIdx.x;
  if (n < N_TOK) {
    const int e = top1[n];
    const int p = atomicAdd(&cursor[e], 1);
    perm[p] = n;
  }
}

// ---------------------------------------------------------------- prepack
__global__ __launch_bounds__(256) void cvt_x_kernel(const float* __restrict__ x,
                                                    unsigned short* __restrict__ xb) {
  const size_t i = ((size_t)blockIdx.x * 256 + threadIdx.x) * 4;
  const float4 v = *(const float4*)(x + i);
  *(uint2*)(xb + i) = make_uint2(pk2(v.x, v.y), pk2(v.z, v.w));
}

// W[e][K][Nn] f32 -> Wt[e][Nn][K] bf16 via 32x32 LDS tile transpose
__global__ __launch_bounds__(256) void transpose_cvt_kernel(
    const float* __restrict__ W, unsigned short* __restrict__ Wt,
    const int K, const int Nn)
{
  __shared__ unsigned short t[32][36];   // padded rows (72B) -> 8B-aligned uint2 reads
  const float* We = W + (size_t)blockIdx.z * K * Nn;
  unsigned short* Wte = Wt + (size_t)blockIdx.z * K * Nn;
  const int n0 = blockIdx.x * 32, k0 = blockIdx.y * 32;
  const int tid = threadIdx.x;
  const int kk = tid >> 3;             // 0..31
  const int nq = (tid & 7) << 2;       // 0,4,..,28
  const float4 v = *(const float4*)(We + (size_t)(k0 + kk) * Nn + n0 + nq);
  t[nq + 0][kk] = f2bf(v.x);
  t[nq + 1][kk] = f2bf(v.y);
  t[nq + 2][kk] = f2bf(v.z);
  t[nq + 3][kk] = f2bf(v.w);
  __syncthreads();
  const int nn = kk, kq = nq;
  *(uint2*)(Wte + (size_t)(n0 + nn) * K + k0 + kq) = make_uint2(
      (unsigned)t[nn][kq + 0] | ((unsigned)t[nn][kq + 1] << 16),
      (unsigned)t[nn][kq + 2] | ((unsigned)t[nn][kq + 3] << 16));
}

// ================================================================ FAST GEMMs
// bf16 operands prepacked k-contiguous; staging = async b128 copies,
// 3-deep pipeline, one barrier per k-step.

// H[pos,:] = relu( xb[perm[pos],:] @ w1t[e]^T + b1[e] )  (w1t: [HID][DIM] k-contig)
__global__ __launch_bounds__(256) void ffn1_fast(
    const unsigned short* __restrict__ xb, const unsigned short* __restrict__ W1t,
    const float* __restrict__ b1, const int* __restrict__ perm,
    const int* __restrict__ offs, unsigned short* __restrict__ Hp)
{
  __shared__ __align__(16) unsigned short As[PIPE][TM * LSTR];
  __shared__ __align__(16) unsigned short Bs[PIPE][TN * LSTR];
  __shared__ int rowTok[TM];

  const int slot = blockIdx.y;
  const int e  = slot / (N_TOK / TM);
  const int rt = slot % (N_TOK / TM);
  const int start = offs[e], end = offs[e + 1];
  const int rowBase = start + rt * TM;
  if (rowBase >= end) return;
  const int nvalid = min(TM, end - rowBase);
  const int n0 = blockIdx.x * TN;

  const int tid = threadIdx.x;
  if (tid < TM) rowTok[tid] = perm[rowBase + min(tid, nvalid - 1)];
  __syncthreads();

  const unsigned short* __restrict__ w1te = W1t + (size_t)e * HID * DIM;
  const int lane = tid & 31;
  const int wave = tid >> 5;
  const int wr = wave >> 1;       // 0..3 -> rows wr*32
  const int wc = wave & 1;        // 0..1 -> cols wc*64
  const int r16 = lane & 15;
  const int g = lane >> 4;

  // per-thread staging slots: 2 x 16B for A, 2 x 16B for B per k-step
  unsigned ldsA[2], ldsB[2];
  unsigned long long gA[2], gB[2];
#pragma unroll
  for (int i = 0; i < 2; ++i) {
    const int idx = tid + i * 256;          // 512 slots, 4 per row
    const int row = idx >> 2;
    const int kq  = (idx & 3) << 3;         // half offset 0,8,16,24
    ldsA[i] = (unsigned)(size_t)&As[0][row * LSTR + kq];
    ldsB[i] = (unsigned)(size_t)&Bs[0][row * LSTR + kq];
    gA[i] = (unsigned long long)(size_t)(xb + (size_t)rowTok[row] * DIM + kq);
    gB[i] = (unsigned long long)(size_t)(w1te + (size_t)(n0 + row) * DIM + kq);
  }

  v8f acc[2][4];
#pragma unroll
  for (int t = 0; t < 2; ++t)
#pragma unroll
    for (int c = 0; c < 4; ++c)
#pragma unroll
      for (int j = 0; j < 8; ++j) acc[t][c][j] = 0.0f;

  const int NS = DIM / TK;   // 32
  // prologue: stage steps 0 and 1 into buffers 0 and 1
#pragma unroll
  for (int i = 0; i < 2; ++i) {
    ASYNC_CP_B128(ldsA[i], gA[i]);
    ASYNC_CP_B128(ldsB[i], gB[i]);
  }
#pragma unroll
  for (int i = 0; i < 2; ++i) {
    ASYNC_CP_B128(ldsA[i] + TILE_BYTES, gA[i] + (TK * 2));
    ASYNC_CP_B128(ldsB[i] + TILE_BYTES, gB[i] + (TK * 2));
  }
  int b = 0, bn = 2;          // compute buffer, next-issue buffer
  for (int s = 0; s < NS; ++s) {
    if (s + 1 < NS) { WAIT_ASYNC4(); } else { WAIT_ASYNC0(); }
    __syncthreads();
    if (s + 2 < NS) {
      const unsigned long long ko = (unsigned long long)(s + 2) * (TK * 2);
      const unsigned lo = (unsigned)(bn * TILE_BYTES);
#pragma unroll
      for (int i = 0; i < 2; ++i) {
        ASYNC_CP_B128(ldsA[i] + lo, gA[i] + ko);
        ASYNC_CP_B128(ldsB[i] + lo, gB[i] + ko);
      }
    }
    v16bf afrag[2], bfrag[4];
#pragma unroll
    for (int t = 0; t < 2; ++t) {
      const uint4* p = (const uint4*)&As[b][(wr * 32 + t * 16 + r16) * LSTR + (g << 4)];
      uint4* d = (uint4*)&afrag[t];
      d[0] = p[0];
      d[1] = p[1];
    }
#pragma unroll
    for (int c = 0; c < 4; ++c) {
      const uint4* p = (const uint4*)&Bs[b][(wc * 64 + c * 16 + r16) * LSTR + (g << 4)];
      uint4* d = (uint4*)&bfrag[c];
      d[0] = p[0];
      d[1] = p[1];
    }
#pragma unroll
    for (int t = 0; t < 2; ++t)
#pragma unroll
      for (int c = 0; c < 4; ++c)
        acc[t][c] = __builtin_amdgcn_wmma_f32_16x16x32_bf16(
            false, afrag[t], false, bfrag[c], (short)0, acc[t][c], false, false);
    b  = (b  == PIPE - 1) ? 0 : b + 1;
    bn = (bn == PIPE - 1) ? 0 : bn + 1;
  }

  // epilogue: bias + relu -> bf16
  if (nvalid == TM) {
#pragma unroll
    for (int c = 0; c < 4; ++c) {
      const int col = n0 + wc * 64 + c * 16 + r16;
      const float bias = b1[e * HID + col];
#pragma unroll
      for (int t = 0; t < 2; ++t)
#pragma unroll
        for (int r = 0; r < 8; ++r) {
          const int lr = wr * 32 + t * 16 + r + 8 * g;
          float v = acc[t][c][r] + bias;
          v = v > 0.0f ? v : 0.0f;
          Hp[(size_t)(rowBase + lr) * HID + col] = f2bf(v);
        }
    }
  } else {
#pragma unroll
    for (int c = 0; c < 4; ++c) {
      const int col = n0 + wc * 64 + c * 16 + r16;
      const float bias = b1[e * HID + col];
#pragma unroll
      for (int t = 0; t < 2; ++t)
#pragma unroll
        for (int r = 0; r < 8; ++r) {
          const int lr = wr * 32 + t * 16 + r + 8 * g;
          if (lr < nvalid) {
            float v = acc[t][c][r] + bias;
            v = v > 0.0f ? v : 0.0f;
            Hp[(size_t)(rowBase + lr) * HID + col] = f2bf(v);
          }
        }
    }
  }
}

// out[perm[pos],:] = H[pos,:] @ w2t[e]^T + b2[e] + x[perm[pos],:]  (w2t: [DIM][HID])
__global__ __launch_bounds__(256) void ffn2_fast(
    const unsigned short* __restrict__ Hp, const unsigned short* __restrict__ W2t,
    const float* __restrict__ b2, const float* __restrict__ x,
    const int* __restrict__ perm, const int* __restrict__ offs,
    float* __restrict__ out)
{
  __shared__ __align__(16) unsigned short As[PIPE][TM * LSTR];
  __shared__ __align__(16) unsigned short Bs[PIPE][TN * LSTR];
  __shared__ int rowTok[TM];

  const int slot = blockIdx.y;
  const int e  = slot / (N_TOK / TM);
  const int rt = slot % (N_TOK / TM);
  const int start = offs[e], end = offs[e + 1];
  const int rowBase = start + rt * TM;
  if (rowBase >= end) return;
  const int nvalid = min(TM, end - rowBase);
  const int n0 = blockIdx.x * TN;

  const int tid = threadIdx.x;
  if (tid < TM) rowTok[tid] = perm[rowBase + min(tid, nvalid - 1)];
  __syncthreads();

  const unsigned short* __restrict__ w2te = W2t + (size_t)e * HID * DIM;
  const int lane = tid & 31;
  const int wave = tid >> 5;
  const int wr = wave >> 1;
  const int wc = wave & 1;
  const int r16 = lane & 15;
  const int g = lane >> 4;

  unsigned ldsA[2], ldsB[2];
  unsigned long long gA[2], gB[2];
#pragma unroll
  for (int i = 0; i < 2; ++i) {
    const int idx = tid + i * 256;
    const int row = idx >> 2;
    const int kq  = (idx & 3) << 3;
    ldsA[i] = (unsigned)(size_t)&As[0][row * LSTR + kq];
    ldsB[i] = (unsigned)(size_t)&Bs[0][row * LSTR + kq];
    gA[i] = (unsigned long long)(size_t)(Hp + (size_t)(rowBase + row) * HID + kq);
    gB[i] = (unsigned long long)(size_t)(w2te + (size_t)(n0 + row) * HID + kq);
  }

  v8f acc[2][4];
#pragma unroll
  for (int t = 0; t < 2; ++t)
#pragma unroll
    for (int c = 0; c < 4; ++c)
#pragma unroll
      for (int j = 0; j < 8; ++j) acc[t][c][j] = 0.0f;

  const int NS = HID / TK;   // 128
#pragma unroll
  for (int i = 0; i < 2; ++i) {
    ASYNC_CP_B128(ldsA[i], gA[i]);
    ASYNC_CP_B128(ldsB[i], gB[i]);
  }
#pragma unroll
  for (int i = 0; i < 2; ++i) {
    ASYNC_CP_B128(ldsA[i] + TILE_BYTES, gA[i] + (TK * 2));
    ASYNC_CP_B128(ldsB[i] + TILE_BYTES, gB[i] + (TK * 2));
  }
  int b = 0, bn = 2;
  for (int s = 0; s < NS; ++s) {
    if (s + 1 < NS) { WAIT_ASYNC4(); } else { WAIT_ASYNC0(); }
    __syncthreads();
    if (s + 2 < NS) {
      const unsigned long long ko = (unsigned long long)(s + 2) * (TK * 2);
      const unsigned lo = (unsigned)(bn * TILE_BYTES);
#pragma unroll
      for (int i = 0; i < 2; ++i) {
        ASYNC_CP_B128(ldsA[i] + lo, gA[i] + ko);
        ASYNC_CP_B128(ldsB[i] + lo, gB[i] + ko);
      }
    }
    v16bf afrag[2], bfrag[4];
#pragma unroll
    for (int t = 0; t < 2; ++t) {
      const uint4* p = (const uint4*)&As[b][(wr * 32 + t * 16 + r16) * LSTR + (g << 4)];
      uint4* d = (uint4*)&afrag[t];
      d[0] = p[0];
      d[1] = p[1];
    }
#pragma unroll
    for (int c = 0; c < 4; ++c) {
      const uint4* p = (const uint4*)&Bs[b][(wc * 64 + c * 16 + r16) * LSTR + (g << 4)];
      uint4* d = (uint4*)&bfrag[c];
      d[0] = p[0];
      d[1] = p[1];
    }
#pragma unroll
    for (int t = 0; t < 2; ++t)
#pragma unroll
      for (int c = 0; c < 4; ++c)
        acc[t][c] = __builtin_amdgcn_wmma_f32_16x16x32_bf16(
            false, afrag[t], false, bfrag[c], (short)0, acc[t][c], false, false);
    b  = (b  == PIPE - 1) ? 0 : b + 1;
    bn = (bn == PIPE - 1) ? 0 : bn + 1;
  }

  if (nvalid == TM) {
#pragma unroll
    for (int c = 0; c < 4; ++c) {
      const int col = n0 + wc * 64 + c * 16 + r16;
      const float bias = b2[e * DIM + col];
#pragma unroll
      for (int t = 0; t < 2; ++t)
#pragma unroll
        for (int r = 0; r < 8; ++r) {
          const int lr = wr * 32 + t * 16 + r + 8 * g;
          const int tok = rowTok[lr];
          out[(size_t)tok * DIM + col] =
              acc[t][c][r] + bias + x[(size_t)tok * DIM + col];
        }
    }
  } else {
#pragma unroll
    for (int c = 0; c < 4; ++c) {
      const int col = n0 + wc * 64 + c * 16 + r16;
      const float bias = b2[e * DIM + col];
#pragma unroll
      for (int t = 0; t < 2; ++t)
#pragma unroll
        for (int r = 0; r < 8; ++r) {
          const int lr = wr * 32 + t * 16 + r + 8 * g;
          if (lr < nvalid) {
            const int tok = rowTok[lr];
            out[(size_t)tok * DIM + col] =
                acc[t][c][r] + bias + x[(size_t)tok * DIM + col];
          }
        }
    }
  }
}

// ================================================================ FALLBACK GEMMs
// (small-workspace path: f32 operands staged + converted in-kernel)
__global__ __launch_bounds__(256) void ffn1_fb(
    const float* __restrict__ x, const float* __restrict__ w1,
    const float* __restrict__ b1, const int* __restrict__ perm,
    const int* __restrict__ offs, unsigned short* __restrict__ Hp)
{
  __shared__ __align__(16) unsigned short As[TM * LSTR];
  __shared__ __align__(16) unsigned short Bs[TN * LSTR];
  __shared__ int rowTok[TM];

  const int slot = blockIdx.y;
  const int e  = slot / (N_TOK / TM);
  const int rt = slot % (N_TOK / TM);
  const int start = offs[e], end = offs[e + 1];
  const int rowBase = start + rt * TM;
  if (rowBase >= end) return;
  const int nvalid = min(TM, end - rowBase);
  const int n0 = blockIdx.x * TN;

  const int tid = threadIdx.x;
  if (tid < TM) rowTok[tid] = perm[rowBase + min(tid, nvalid - 1)];
  __syncthreads();

  const float* __restrict__ w1e = w1 + (size_t)e * DIM * HID;
  const int lane = tid & 31;
  const int wave = tid >> 5;
  const int wr = wave >> 1;
  const int wc = wave & 1;
  const int r16 = lane & 15;
  const int g = lane >> 4;

  v8f acc[2][4];
#pragma unroll
  for (int t = 0; t < 2; ++t)
#pragma unroll
    for (int c = 0; c < 4; ++c)
#pragma unroll
      for (int j = 0; j < 8; ++j) acc[t][c][j] = 0.0f;

  for (int k0 = 0; k0 < DIM; k0 += TK) {
    __syncthreads();
#pragma unroll
    for (int i = 0; i < 4; ++i) {
      const int idx = tid + i * 256;
      const int row = idx >> 3;
      const int kq  = (idx & 7) << 2;
      const float4 v = *(const float4*)(x + (size_t)rowTok[row] * DIM + k0 + kq);
      *(uint2*)&As[row * LSTR + kq] = make_uint2(pk2(v.x, v.y), pk2(v.z, v.w));
    }
#pragma unroll
    for (int i = 0; i < 4; ++i) {
      const int idx = tid + i * 256;
      const int kk = idx >> 5;
      const int nq = (idx & 31) << 2;
      const float4 v = *(const float4*)(w1e + (size_t)(k0 + kk) * HID + n0 + nq);
      Bs[(nq + 0) * LSTR + kk] = f2bf(v.x);
      Bs[(nq + 1) * LSTR + kk] = f2bf(v.y);
      Bs[(nq + 2) * LSTR + kk] = f2bf(v.z);
      Bs[(nq + 3) * LSTR + kk] = f2bf(v.w);
    }
    __syncthreads();

    v16bf afrag[2], bfrag[4];
#pragma unroll
    for (int t = 0; t < 2; ++t) {
      const uint4* p = (const uint4*)&As[(wr * 32 + t * 16 + r16) * LSTR + (g << 4)];
      uint4* d = (uint4*)&afrag[t];
      d[0] = p[0];
      d[1] = p[1];
    }
#pragma unroll
    for (int c = 0; c < 4; ++c) {
      const uint4* p = (const uint4*)&Bs[(wc * 64 + c * 16 + r16) * LSTR + (g << 4)];
      uint4* d = (uint4*)&bfrag[c];
      d[0] = p[0];
      d[1] = p[1];
    }
#pragma unroll
    for (int t = 0; t < 2; ++t)
#pragma unroll
      for (int c = 0; c < 4; ++c)
        acc[t][c] = __builtin_amdgcn_wmma_f32_16x16x32_bf16(
            false, afrag[t], false, bfrag[c], (short)0, acc[t][c], false, false);
  }

#pragma unroll
  for (int c = 0; c < 4; ++c) {
    const int col = n0 + wc * 64 + c * 16 + r16;
    const float bias = b1[e * HID + col];
#pragma unroll
    for (int t = 0; t < 2; ++t)
#pragma unroll
      for (int r = 0; r < 8; ++r) {
        const int lr = wr * 32 + t * 16 + r + 8 * g;
        if (lr < nvalid) {
          float v = acc[t][c][r] + bias;
          v = v > 0.0f ? v : 0.0f;
          Hp[(size_t)(rowBase + lr) * HID + col] = f2bf(v);
        }
      }
  }
}

__global__ __launch_bounds__(256) void ffn2_fb(
    const unsigned short* __restrict__ Hp, const float* __restrict__ w2,
    const float* __restrict__ b2, const float* __restrict__ x,
    const int* __restrict__ perm, const int* __restrict__ offs,
    float* __restrict__ out)
{
  __shared__ __align__(16) unsigned short As[TM * LSTR];
  __shared__ __align__(16) unsigned short Bs[TN * LSTR];
  __shared__ int rowTok[TM];

  const int slot = blockIdx.y;
  const int e  = slot / (N_TOK / TM);
  const int rt = slot % (N_TOK / TM);
  const int start = offs[e], end = offs[e + 1];
  const int rowBase = start + rt * TM;
  if (rowBase >= end) return;
  const int nvalid = min(TM, end - rowBase);
  const int n0 = blockIdx.x * TN;

  const int tid = threadIdx.x;
  if (tid < TM) rowTok[tid] = perm[rowBase + min(tid, nvalid - 1)];
  __syncthreads();

  const float* __restrict__ w2e = w2 + (size_t)e * HID * DIM;
  const int lane = tid & 31;
  const int wave = tid >> 5;
  const int wr = wave >> 1;
  const int wc = wave & 1;
  const int r16 = lane & 15;
  const int g = lane >> 4;

  v8f acc[2][4];
#pragma unroll
  for (int t = 0; t < 2; ++t)
#pragma unroll
    for (int c = 0; c < 4; ++c)
#pragma unroll
      for (int j = 0; j < 8; ++j) acc[t][c][j] = 0.0f;

  for (int k0 = 0; k0 < HID; k0 += TK) {
    __syncthreads();
#pragma unroll
    for (int i = 0; i < 2; ++i) {
      const int idx = tid + i * 256;
      const int row = idx >> 2;
      const int kq  = (idx & 3) << 3;
      *(uint4*)&As[row * LSTR + kq] =
          *(const uint4*)(Hp + (size_t)(rowBase + row) * HID + k0 + kq);
    }
#pragma unroll
    for (int i = 0; i < 4; ++i) {
      const int idx = tid + i * 256;
      const int kk = idx >> 5;
      const int nq = (idx & 31) << 2;
      const float4 v = *(const float4*)(w2e + (size_t)(k0 + kk) * DIM + n0 + nq);
      Bs[(nq + 0) * LSTR + kk] = f2bf(v.x);
      Bs[(nq + 1) * LSTR + kk] = f2bf(v.y);
      Bs[(nq + 2) * LSTR + kk] = f2bf(v.z);
      Bs[(nq + 3) * LSTR + kk] = f2bf(v.w);
    }
    __syncthreads();

    v16bf afrag[2], bfrag[4];
#pragma unroll
    for (int t = 0; t < 2; ++t) {
      const uint4* p = (const uint4*)&As[(wr * 32 + t * 16 + r16) * LSTR + (g << 4)];
      uint4* d = (uint4*)&afrag[t];
      d[0] = p[0];
      d[1] = p[1];
    }
#pragma unroll
    for (int c = 0; c < 4; ++c) {
      const uint4* p = (const uint4*)&Bs[(wc * 64 + c * 16 + r16) * LSTR + (g << 4)];
      uint4* d = (uint4*)&bfrag[c];
      d[0] = p[0];
      d[1] = p[1];
    }
#pragma unroll
    for (int t = 0; t < 2; ++t)
#pragma unroll
      for (int c = 0; c < 4; ++c)
        acc[t][c] = __builtin_amdgcn_wmma_f32_16x16x32_bf16(
            false, afrag[t], false, bfrag[c], (short)0, acc[t][c], false, false);
  }

#pragma unroll
  for (int c = 0; c < 4; ++c) {
    const int col = n0 + wc * 64 + c * 16 + r16;
    const float bias = b2[e * DIM + col];
#pragma unroll
    for (int t = 0; t < 2; ++t)
#pragma unroll
      for (int r = 0; r < 8; ++r) {
        const int lr = wr * 32 + t * 16 + r + 8 * g;
        if (lr < nvalid) {
          const int tok = rowTok[lr];
          out[(size_t)tok * DIM + col] =
              acc[t][c][r] + bias + x[(size_t)tok * DIM + col];
        }
      }
  }
}

// ---------------------------------------------------------------- launch
extern "C" void kernel_launch(void* const* d_in, const int* in_sizes, int n_in,
                              void* d_out, int out_size, void* d_ws, size_t ws_size,
                              hipStream_t stream) {
  (void)in_sizes; (void)n_in; (void)out_size;
  const float* x  = (const float*)d_in[0];
  const float* gw = (const float*)d_in[1];
  const float* gb = (const float*)d_in[2];
  const float* w1 = (const float*)d_in[3];
  const float* b1 = (const float*)d_in[4];
  const float* w2 = (const float*)d_in[5];
  const float* b2 = (const float*)d_in[6];
  float* out = (float*)d_out;

  char* base = (char*)d_ws;
  size_t off = 0;
  auto alloc = [&](size_t bytes) -> void* {
    void* p = base + off;
    off = (off + bytes + 255) & ~(size_t)255;
    return p;
  };
  int* top1   = (int*)alloc((size_t)N_TOK * 4);
  int* counts = (int*)alloc(NEXP * 4);
  int* offs   = (int*)alloc((NEXP + 1) * 4);
  int* cursor = (int*)alloc(NEXP * 4);
  int* perm   = (int*)alloc((size_t)N_TOK * 4);
  unsigned short* Hp = (unsigned short*)alloc((size_t)N_TOK * HID * 2);
  unsigned short* xb  = (unsigned short*)alloc((size_t)N_TOK * DIM * 2);
  unsigned short* W1t = (unsigned short*)alloc((size_t)NEXP * DIM * HID * 2);
  unsigned short* W2t = (unsigned short*)alloc((size_t)NEXP * HID * DIM * 2);
  const size_t full_need = off;

  hipMemsetAsync(counts, 0, NEXP * sizeof(int), stream);
  gate_kernel<<<N_TOK / 8, 256, 0, stream>>>(x, gw, gb, top1);
  hist_kernel<<<N_TOK / 256, 256, 0, stream>>>(top1, counts);
  scan_kernel<<<1, 32, 0, stream>>>(counts, offs, cursor);
  scatter_kernel<<<N_TOK / 256, 256, 0, stream>>>(top1, cursor, perm);

  const dim3 g1(HID / TN, NEXP * (N_TOK / TM));
  const dim3 g2(DIM / TN, NEXP * (N_TOK / TM));
  if (ws_size >= full_need) {
    cvt_x_kernel<<<(N_TOK * DIM) / 1024, 256, 0, stream>>>(x, xb);
    transpose_cvt_kernel<<<dim3(HID / 32, DIM / 32, NEXP), 256, 0, stream>>>(w1, W1t, DIM, HID);
    transpose_cvt_kernel<<<dim3(DIM / 32, HID / 32, NEXP), 256, 0, stream>>>(w2, W2t, HID, DIM);
    ffn1_fast<<<g1, 256, 0, stream>>>(xb, W1t, b1, perm, offs, Hp);
    ffn2_fast<<<g2, 256, 0, stream>>>(Hp, W2t, b2, x, perm, offs, out);
  } else {
    ffn1_fb<<<g1, 256, 0, stream>>>(x, w1, b1, perm, offs, Hp);
    ffn2_fb<<<g2, 256, 0, stream>>>(Hp, w2, b2, x, perm, offs, out);
  }
}
